// CosineSimilarityModel_64570538328854
// MI455X (gfx1250) — matrix operand order repaired and checked by reading, not compiled, not run on hardware
//
#include <hip/hip_runtime.h>
#include <hip/hip_bf16.h>

// ---- CDNA5 (gfx1250) wave32 WMMA GEMM for cosine-similarity model ----
// out[B,462] = l2norm_rows(x[B,512]) @ l2norm_rows(W[512,462])

typedef _Float16 h16 __attribute__((ext_vector_type(16)));
typedef _Float16 h8  __attribute__((ext_vector_type(8)));
typedef _Float16 h4  __attribute__((ext_vector_type(4)));
typedef float    f8  __attribute__((ext_vector_type(8)));

#define IN_DIM  512
#define OUT_DIM 462
#define NPAD    512        // Bt padded N so every N-block load is in-bounds
#define BM      128
#define BN      64
#define BK      32
#define ASTRIDE (BK + 8)   // 40 halves -> 80B row stride, conflict-free b128 reads

// ---------------- Kernel 1: per-row inverse L2 norm of x ----------------
__global__ __launch_bounds__(256) void xnorm_kernel(const float* __restrict__ x,
                                                    float* __restrict__ invn,
                                                    int rows) {
    int row  = blockIdx.x * 8 + (threadIdx.x >> 5);
    int lane = threadIdx.x & 31;
    if (row >= rows) return;
    const float4* xr = (const float4*)(x + (size_t)row * IN_DIM);
    float s = 0.f;
#pragma unroll
    for (int c = 0; c < 4; ++c) {
        float4 v = xr[lane + 32 * c];
        s += v.x * v.x + v.y * v.y + v.z * v.z + v.w * v.w;
    }
#pragma unroll
    for (int m = 16; m >= 1; m >>= 1) s += __shfl_xor(s, m, 32);
    if (lane == 0) invn[row] = rsqrtf(fmaxf(s, 1e-12f));
}

// ------- Kernel 2: normalize W rows (over N) -> f16, transposed + padded -------
// Bt[n][k] = W[k][n] / ||W[k,:]||, zero for n in [462,512)
__global__ __launch_bounds__(256) void wprep_kernel(const float* __restrict__ W,
                                                    _Float16* __restrict__ Bt) {
    int k    = blockIdx.x * 8 + (threadIdx.x >> 5);   // 0..511
    int lane = threadIdx.x & 31;
    const float* wr = W + (size_t)k * OUT_DIM;
    float s = 0.f;
    for (int c = lane; c < OUT_DIM; c += 32) { float v = wr[c]; s += v * v; }
#pragma unroll
    for (int m = 16; m >= 1; m >>= 1) s += __shfl_xor(s, m, 32);
    float inv = rsqrtf(fmaxf(s, 1e-12f));
    for (int c = lane; c < NPAD; c += 32) {
        float v = (c < OUT_DIM) ? wr[c] * inv : 0.f;
        Bt[(size_t)c * IN_DIM + k] = (_Float16)v;
    }
}

// ---------------- Kernel 3: WMMA GEMM (f16 inputs, f32 accumulate) ----------------
__global__ __launch_bounds__(256) void gemm_kernel(const float* __restrict__ x,
                                                   const float* __restrict__ invn,
                                                   const _Float16* __restrict__ Bt,
                                                   float* __restrict__ out) {
    __shared__ _Float16 As[2][BM * ASTRIDE];
    __shared__ float    invs[BM];

    const int tid   = threadIdx.x;
    const int lane  = tid & 31;
    const int wid   = tid >> 5;
    const int waveM = wid & 3;    // 4 M-groups of 32 rows
    const int waveN = wid >> 2;   // 2 N-groups of 32 cols
    const int g     = lane >> 4;  // half-wave group
    const int ln    = lane & 15;

    const int mBase = blockIdx.y * BM;
    const int nBase = blockIdx.x * BN;

    // staging decode: 1024 float4 per tile, 4 per thread
    const int srow = tid >> 3;           // 0..31 (+32*j)
    const int scol = (tid & 7) << 2;     // 0,4,...,28

    if (tid < BM) invs[tid] = invn[mBase + tid];

    f8 acc00 = {}, acc01 = {}, acc10 = {}, acc11 = {};

    float4 regs[4];
    // prologue: global load of k-tile 0
#pragma unroll
    for (int j = 0; j < 4; ++j)
        regs[j] = *(const float4*)(x + (size_t)(mBase + srow + 32 * j) * IN_DIM + scol);
    __syncthreads();   // invs visible
#pragma unroll
    for (int j = 0; j < 4; ++j) {
        float sc = invs[srow + 32 * j];
        h4 hv = { (_Float16)(regs[j].x * sc), (_Float16)(regs[j].y * sc),
                  (_Float16)(regs[j].z * sc), (_Float16)(regs[j].w * sc) };
        *(h4*)&As[0][(srow + 32 * j) * ASTRIDE + scol] = hv;
    }
    __syncthreads();   // tile 0 visible

    int buf = 0;
    for (int kt = 0; kt < IN_DIM / BK; ++kt) {
        // software pipeline: fetch next f32 tile while we compute
        if (kt + 1 < IN_DIM / BK) {
#pragma unroll
            for (int j = 0; j < 4; ++j)
                regs[j] = *(const float4*)(x + (size_t)(mBase + srow + 32 * j) * IN_DIM +
                                           (kt + 1) * BK + scol);
        }

        // --- compute on buffer `buf` ---
        const _Float16* Ab  = As[buf];
        const _Float16* a0p = Ab + (waveM * 32 + ln) * ASTRIDE + 8 * g;
        const _Float16* a1p = a0p + 16 * ASTRIDE;
        h16 a0, a1;
        *((h8*)&a0)     = *(const h8*)(a0p);
        *((h8*)&a0 + 1) = *(const h8*)(a0p + 16);
        *((h8*)&a1)     = *(const h8*)(a1p);
        *((h8*)&a1 + 1) = *(const h8*)(a1p + 16);

        const _Float16* bp = Bt + (size_t)(nBase + waveN * 32 + ln) * IN_DIM + kt * BK + 8 * g;
        h16 b0, b1;
        *((h8*)&b0)     = *(const h8*)(bp);
        *((h8*)&b0 + 1) = *(const h8*)(bp + 16);
        *((h8*)&b1)     = *(const h8*)(bp + (size_t)16 * IN_DIM);
        *((h8*)&b1 + 1) = *(const h8*)(bp + (size_t)16 * IN_DIM + 16);

        acc00 = __builtin_amdgcn_wmma_f32_16x16x32_f16(false, a0, false, b0, (short)0, acc00, false, false);
        acc01 = __builtin_amdgcn_wmma_f32_16x16x32_f16(false, a0, false, b1, (short)0, acc01, false, false);
        acc10 = __builtin_amdgcn_wmma_f32_16x16x32_f16(false, a1, false, b0, (short)0, acc10, false, false);
        acc11 = __builtin_amdgcn_wmma_f32_16x16x32_f16(false, a1, false, b1, (short)0, acc11, false, false);

        // --- stage next tile into other buffer ---
        if (kt + 1 < IN_DIM / BK) {
#pragma unroll
            for (int j = 0; j < 4; ++j) {
                float sc = invs[srow + 32 * j];
                h4 hv = { (_Float16)(regs[j].x * sc), (_Float16)(regs[j].y * sc),
                          (_Float16)(regs[j].z * sc), (_Float16)(regs[j].w * sc) };
                *(h4*)&As[buf ^ 1][(srow + 32 * j) * ASTRIDE + scol] = hv;
            }
        }
        __syncthreads();
        buf ^= 1;
    }

    // --- store: C layout = VGPR r -> row r + 8*g, lane&15 -> column; NT to spare L2 ---
    const int colBase = nBase + waveN * 32 + ln;
    const int rowBase = mBase + waveM * 32 + 8 * g;
#pragma unroll
    for (int subn = 0; subn < 2; ++subn) {
        int col = colBase + subn * 16;
        if (col >= OUT_DIM) continue;
#pragma unroll
        for (int r = 0; r < 8; ++r) {
            __builtin_nontemporal_store(subn ? acc01[r] : acc00[r],
                out + (size_t)(rowBase + r) * OUT_DIM + col);
            __builtin_nontemporal_store(subn ? acc11[r] : acc10[r],
                out + (size_t)(rowBase + 16 + r) * OUT_DIM + col);
        }
    }
}

extern "C" void kernel_launch(void* const* d_in, const int* in_sizes, int n_in,
                              void* d_out, int out_size, void* d_ws, size_t ws_size,
                              hipStream_t stream) {
    const float* x = (const float*)d_in[0];
    const float* W = (const float*)d_in[1];
    float* out = (float*)d_out;
    const int rows = in_sizes[0] / IN_DIM;   // 65536

    float*    invn = (float*)d_ws;
    size_t    off  = (((size_t)rows * sizeof(float)) + 255) & ~(size_t)255;
    _Float16* Bt   = (_Float16*)((char*)d_ws + off);   // [NPAD][IN_DIM] f16

    xnorm_kernel<<<rows / 8, 256, 0, stream>>>(x, invn, rows);
    wprep_kernel<<<IN_DIM / 8, 256, 0, stream>>>(W, Bt);
    gemm_kernel<<<dim3((NPAD + BN - 1) / BN, rows / BM), 256, 0, stream>>>(x, invn, Bt, out);
}
